// Seq2Seq_22153441313116
// MI455X (gfx1250) — matrix operand order repaired and checked by reading, not compiled, not run on hardware
//
#include <hip/hip_runtime.h>

// ---------------- problem constants (match reference) ----------------
#define HID    10     // hidden dim
#define OUTN   100    // charset logits
#define DSTEPS 30     // decoder steps
#define ESTEPS 32     // encoder steps (L)
#define TD     63     // tag dim
#define INDIM  64     // 1 char + 63 tags
#define BATCH  8192
#define STARTV 1.0f

typedef _Float16 f16;
typedef __attribute__((ext_vector_type(16))) _Float16 v16h;
typedef __attribute__((ext_vector_type(8)))  float    v8f;

// ---------------- LDS layout (bytes) ----------------
// packed f16 weights (halves):
//  enc0 @0 (4 tiles *3 chunks *16cols *32k = 6144)   enc1 @6144 (2048)  enc2 @8192 (2048)
//  dec0 @10240 (6144)  dec1 @16384 (2048)  dec2 @18432 (2048)  lin @20480 (7*512=3584)
//  total 24064 halves = 48128 B
#define BIAS_OFF    48128              // 6*64 + 112 = 496 floats = 1984 B
#define STG_OFF     50112              // per-wave staging base
#define STG_PER_W   13312              // x0(3072) x1(1024) x2(1024) lem(1024) pbuf(7168)
#define WAVES       2
#define SMEM_BYTES  (STG_OFF + WAVES * STG_PER_W)   // 76736

struct SeqParams {
    const float* tags;          // (B,63)
    const float* lemma;         // (B,32)
    const float* Wih[6];        // enc0..2, dec0..2  (4H, din)
    const float* Whh[6];        // (4H, H)
    const float* bih[6];
    const float* bhh[6];
    const float* linW;          // (100, H)
    const float* linb;          // (100,)
    float* out;                 // (B,30,100)
    float* tok;                 // (B,30) as float
};

// v_exp_f32 + v_rcp_f32 (both TRANS, co-issue with WMMA) instead of IEEE divide expansion
__device__ __forceinline__ float sigm(float x) {
    return __builtin_amdgcn_rcpf(1.f + __expf(-x));
}
__device__ __forceinline__ float tanha(float x) {
    return 2.f * __builtin_amdgcn_rcpf(1.f + __expf(-2.f * x)) - 1.f;
}

// A-fragment (16x32 f16, MxK): lane<16 holds row=lane, K 0..15;
// lane>=16: same rows, K 16..31. Two 16B LDS loads.
__device__ __forceinline__ v16h ld_a(const f16* xb, int xs, int lane) {
    const int row = lane & 15;
    const int kb  = (lane >> 4) << 4;
    const f16* p = xb + row * xs + kb;
    v16h a;
    *(uint4*)&a        = *(const uint4*)(p);
    *((uint4*)&a + 1)  = *(const uint4*)(p + 8);
    return a;
}

// B-fragment (32x16 f16, KxN): lane<16 = column lane, K 0..15; lane>=16 = column lane-16, K 16..31.
// Weights stored column-major per 32-K chunk -> 16 consecutive halves per lane.
__device__ __forceinline__ v16h ld_b(const f16* wb, int lane) {
    const int col = lane & 15;
    const int k0  = (lane >> 4) << 4;
    const f16* p = wb + col * 32 + k0;
    v16h b;
    *(uint4*)&b        = *(const uint4*)(p);
    *((uint4*)&b + 1)  = *(const uint4*)(p + 8);
    return b;
}

// One LSTM layer step for a 16-batch tile.
// xb: concat input buffer [16 rows][KC*32] f16 (K = [input | h_prev] padded).
// wu: packed weights, 4 gate tiles (i,f,g,o) x KC chunks x (16 col x 32 k).
// C/D layout: lane holds column (= hidden unit, valid < 10), 8 VGPRs = 8 batch rows.
// h (f16) is written to d0 (next-layer input cols) and d1 (own recurrent cols).
template<int KC>
__device__ __forceinline__ void lstm_step(const f16* xb, int xs, const f16* wu, const float* bu,
                                          v8f& cst, f16* d0, int s0, f16* d1, int s1, int lane) {
    const int col = lane & 15;
    const int grp = lane >> 4;
    const float b0 = bu[col], b1 = bu[16 + col], b2 = bu[32 + col], b3 = bu[48 + col];
    v8f a0, a1, a2, a3;
#pragma unroll
    for (int r = 0; r < 8; ++r) { a0[r] = b0; a1[r] = b1; a2[r] = b2; a3[r] = b3; }
#pragma unroll
    for (int ch = 0; ch < KC; ++ch) {
        // preload A + all 4 gate B-fragments, then issue WMMAs back-to-back
        // (partial dscnt waits instead of a full drain before every WMMA)
        const v16h av = ld_a(xb + ch * 32, xs, lane);
        const v16h w0 = ld_b(wu + (0 * KC + ch) * 512, lane);
        const v16h w1 = ld_b(wu + (1 * KC + ch) * 512, lane);
        const v16h w2 = ld_b(wu + (2 * KC + ch) * 512, lane);
        const v16h w3 = ld_b(wu + (3 * KC + ch) * 512, lane);
        a0 = __builtin_amdgcn_wmma_f32_16x16x32_f16(false, av, false, w0, (short)0, a0, false, false);
        a1 = __builtin_amdgcn_wmma_f32_16x16x32_f16(false, av, false, w1, (short)0, a1, false, false);
        a2 = __builtin_amdgcn_wmma_f32_16x16x32_f16(false, av, false, w2, (short)0, a2, false, false);
        a3 = __builtin_amdgcn_wmma_f32_16x16x32_f16(false, av, false, w3, (short)0, a3, false, false);
    }
    v8f hn;
#pragma unroll
    for (int r = 0; r < 8; ++r) {
        const float ig = sigm(a0[r]);
        const float fg = sigm(a1[r]);
        const float gg = tanha(a2[r]);
        const float og = sigm(a3[r]);
        const float cn = fg * cst[r] + ig * gg;
        cst[r] = cn;
        hn[r]  = og * tanha(cn);
    }
    if (col < HID) {
#pragma unroll
        for (int r = 0; r < 8; ++r) {
            const f16 hv = (f16)hn[r];
            const int row = (grp << 3) + r;
            if (d0) d0[row * s0 + col] = hv;
            d1[row * s1 + col] = hv;
        }
    }
}

__global__ __launch_bounds__(32 * WAVES) void seq2seq_kernel(SeqParams P) {
    extern __shared__ char smem[];
    f16*   wp   = (f16*)smem;
    float* bias = (float*)(smem + BIAS_OFF);

    const int lane = threadIdx.x & 31;
    const int wave = threadIdx.x >> 5;

    // warm L2 for the weight streams while we run the packing stage (global_prefetch_b8)
#pragma unroll
    for (int u = 0; u < 6; ++u) { __builtin_prefetch(P.Wih[u], 0, 3); __builtin_prefetch(P.Whh[u], 0, 3); }
    __builtin_prefetch(P.linW, 0, 3);

    // -------- stage 0: pack f32 weights -> f16 WMMA-B fragments in LDS --------
    const int UOFF[6] = {0, 6144, 8192, 10240, 16384, 18432};
#pragma unroll
    for (int u = 0; u < 6; ++u) {
        const int layer = u % 3;
        const int din   = layer ? HID : INDIM;
        const int KC    = layer ? 1 : 3;
        const int total = 4 * KC * 512;
        const float* wih = P.Wih[u];
        const float* whh = P.Whh[u];
        f16* wb = wp + UOFF[u];
        for (int idx = threadIdx.x; idx < total; idx += blockDim.x) {
            const int k  = idx & 31;
            const int co = (idx >> 5) & 15;
            const int tc = idx >> 9;
            const int ch = tc % KC, tl = tc / KC;       // tl = gate type (i,f,g,o)
            const int row = tl * HID + co;              // W row for gate tl, hidden unit co
            const int kk  = ch * 32 + k;                // concat-K: [input(din) | h(10) | pad]
            float v = 0.f;
            if (co < HID) {
                if (kk < din)            v = wih[row * din + kk];
                else if (kk < din + HID) v = whh[row * HID + (kk - din)];
            }
            wb[idx] = (f16)v;
        }
        for (int idx = threadIdx.x; idx < 64; idx += blockDim.x) {
            const int co = idx & 15, tl = idx >> 4, row = tl * HID + co;
            bias[u * 64 + idx] = (co < HID) ? (P.bih[u][row] + P.bhh[u][row]) : 0.f;
        }
    }
    // linear head: packed at K offset 10 so the layer-2 concat buffer is reused as A
    for (int idx = threadIdx.x; idx < 3584; idx += blockDim.x) {
        const int k = idx & 31, co = (idx >> 5) & 15, tl = idx >> 9;
        const int n = tl * 16 + co;
        float v = 0.f;
        if (n < OUTN && k >= HID && k < 2 * HID) v = P.linW[n * HID + (k - HID)];
        wp[20480 + idx] = (f16)v;
    }
    for (int idx = threadIdx.x; idx < 112; idx += blockDim.x)
        bias[384 + idx] = (idx < OUTN) ? P.linb[idx] : 0.f;
    __syncthreads();

    // -------- per-wave staging: one wave = one 16-row batch tile --------
    char*  stg = smem + STG_OFF + wave * STG_PER_W;
    f16*   x0  = (f16*)stg;                  // [16][96]: col0=char, 1..63=tags, 64..73=h0, rest 0
    f16*   x1  = (f16*)(stg + 3072);         // [16][32]: 0..9=h0(cur), 10..19=h1(prev), rest 0
    f16*   x2  = (f16*)(stg + 4096);         // [16][32]: 0..9=h1(cur), 10..19=h2(prev), rest 0
    f16*   lem = (f16*)(stg + 5120);         // [16][32]: pre-converted lemma chars
    float* pb  = (float*)(stg + 6144);       // [16][112] probs for argmax
    const int tileBase = (blockIdx.x * WAVES + wave) * 16;

    { f16* z = (f16*)stg; for (int i = lane; i < 2560; i += 32) z[i] = (f16)0.f; }
    for (int i = lane; i < 16 * TD; i += 32) {
        const int m = i / TD, c = i % TD;
        x0[m * 96 + 1 + c] = (f16)P.tags[(size_t)(tileBase + m) * TD + c];
    }
    for (int i = lane; i < 16 * ESTEPS; i += 32)
        lem[i] = (f16)P.lemma[(size_t)tileBase * ESTEPS + i];

    const int col = lane & 15, grp = lane >> 4;

    v8f c0, c1, c2;
#pragma unroll
    for (int r = 0; r < 8; ++r) { c0[r] = 0.f; c1[r] = 0.f; c2[r] = 0.f; }

    const f16 *WE0 = wp, *WE1 = wp + 6144, *WE2 = wp + 8192;
    const f16 *WD0 = wp + 10240, *WD1 = wp + 16384, *WD2 = wp + 18432;
    const f16 *WLN = wp + 20480;

    // -------- encoder --------
    for (int t = 0; t < ESTEPS; ++t) {
        if (lane < 16) x0[lane * 96] = lem[lane * ESTEPS + t];
        lstm_step<3>(x0, 96, WE0, bias +   0, c0, x1, 32, x0 + 64, 96, lane);
        lstm_step<1>(x1, 32, WE1, bias +  64, c1, x2, 32, x1 + 10, 32, lane);
        lstm_step<1>(x2, 32, WE2, bias + 128, c2, (f16*)nullptr, 0, x2 + 10, 32, lane);
    }

    // -------- decoder (argmax feedback), states carry over from encoder --------
    float tokf = STARTV;    // lanes 0..15 hold their row's token
    for (int t = 0; t < DSTEPS; ++t) {
        if (lane < 16) x0[lane * 96] = (f16)tokf;
        lstm_step<3>(x0, 96, WD0, bias + 192, c0, x1, 32, x0 + 64, 96, lane);
        lstm_step<1>(x1, 32, WD1, bias + 256, c1, x2, 32, x1 + 10, 32, lane);
        lstm_step<1>(x2, 32, WD2, bias + 320, c2, (f16*)nullptr, 0, x2 + 10, 32, lane);

        const v16h av = ld_a(x2, 32, lane);     // cols 10..19 = h2, matched by lin packing
#pragma unroll 2
        for (int tl = 0; tl < 7; ++tl) {
            const v16h wl = ld_b(WLN + tl * 512, lane);
            const float bb = bias[384 + tl * 16 + col];
            v8f acc;
#pragma unroll
            for (int r = 0; r < 8; ++r) acc[r] = bb;
            acc = __builtin_amdgcn_wmma_f32_16x16x32_f16(false, av, false, wl, (short)0, acc, false, false);
            const int n = tl * 16 + col;
#pragma unroll
            for (int r = 0; r < 8; ++r) {
                const float p = sigm(acc[r]);
                const int row = (grp << 3) + r;
                pb[row * 112 + n] = p;
                if (n < OUTN)
                    P.out[((size_t)(tileBase + row) * DSTEPS + t) * OUTN + n] = p;
            }
        }
        if (lane < 16) {
            const float* pr = pb + lane * 112;
            float best = pr[0]; int bi = 0;
#pragma unroll 4
            for (int n = 1; n < OUTN; ++n) { const float v = pr[n]; if (v > best) { best = v; bi = n; } }
            tokf = (float)bi;
            P.tok[(size_t)(tileBase + lane) * DSTEPS + t] = (float)bi;
        }
    }
}

extern "C" void kernel_launch(void* const* d_in, const int* in_sizes, int n_in,
                              void* d_out, int out_size, void* d_ws, size_t ws_size,
                              hipStream_t stream) {
    (void)in_sizes; (void)n_in; (void)d_ws; (void)ws_size; (void)out_size;
    SeqParams P;
    P.tags  = (const float*)d_in[2];
    P.lemma = (const float*)d_in[3];
    for (int u = 0; u < 6; ++u) {           // enc0..2 then dec0..2
        P.Wih[u] = (const float*)d_in[4 + 4 * u];
        P.Whh[u] = (const float*)d_in[5 + 4 * u];
        P.bih[u] = (const float*)d_in[6 + 4 * u];
        P.bhh[u] = (const float*)d_in[7 + 4 * u];
    }
    P.linW = (const float*)d_in[28];
    P.linb = (const float*)d_in[29];
    P.out  = (float*)d_out;
    P.tok  = (float*)d_out + (size_t)BATCH * DSTEPS * OUTN;

    hipFuncSetAttribute((const void*)seq2seq_kernel,
                        hipFuncAttributeMaxDynamicSharedMemorySize, SMEM_BYTES);
    seq2seq_kernel<<<dim3(BATCH / (16 * WAVES)), dim3(32 * WAVES), SMEM_BYTES, stream>>>(P);
}
